// GSDecoder_20529943675441
// MI455X (gfx1250) — matrix operand order repaired and checked by reading, not compiled, not run on hardware
//
#include <hip/hip_runtime.h>
#include <hip/hip_bf16.h>
#include <math.h>

// ---------------------------------------------------------------------------
// Types
// ---------------------------------------------------------------------------
typedef _Float16 half_t;
typedef __attribute__((ext_vector_type(16))) _Float16 v16h;
typedef __attribute__((ext_vector_type(8)))  _Float16 v8h;
typedef __attribute__((ext_vector_type(8)))  float    v8f;

#define N_TOK 4096
#define C_DIM 512
#define L_NUM 8
#define H_NUM 8
#define DH    64
#define WS    512
#define NW3   8
#define MLP_D 2048
#define OUTC  56
#define NG    4

// ---------------------------------------------------------------------------
// WMMA helpers (CDNA5 wave32, 16x16x32 f16 -> f32)
// ---------------------------------------------------------------------------
__device__ __forceinline__ v8f wmma32(v16h a, v16h b, v8f c) {
  return __builtin_amdgcn_wmma_f32_16x16x32_f16(
      /*neg_a=*/false, a, /*neg_b=*/false, b,
      /*c_mod=*/(short)0, c, /*reuse_a=*/false, /*reuse_b=*/false);
}

// A fragment 16x32 f16: lane l holds row (l&15); halves: lanes<16 -> K 0..7 &
// 16..23, lanes>=16 -> K 8..15 & 24..31 (two b128 loads).
__device__ __forceinline__ v16h frag_a(const half_t* p0, int stride, int lane) {
  const half_t* p = p0 + (size_t)(lane & 15) * stride + ((lane >> 4) << 3);
  v8h lo = *(const v8h*)(p);
  v8h hi = *(const v8h*)(p + 16);
  v16h r;
#pragma unroll
  for (int i = 0; i < 8; ++i) { r[i] = lo[i]; r[i + 8] = hi[i]; }
  return r;
}

// B fragment 32x16 f16 (stored column-major as Bt[n][k]): lane l holds column
// (l&15); lanes<16 -> K 0..7 & 16..23, lanes>=16 -> K 8..15 & 24..31 (two b128 loads).
__device__ __forceinline__ v16h frag_b(const half_t* p0, int stride, int lane) {
  const half_t* p = p0 + (size_t)(lane & 15) * stride + ((lane >> 4) << 3);
  v8h lo = *(const v8h*)(p);
  v8h hi = *(const v8h*)(p + 16);
  v16h r;
#pragma unroll
  for (int i = 0; i < 8; ++i) { r[i] = lo[i]; r[i + 8] = hi[i]; }
  return r;
}

__device__ __forceinline__ float gelu_f(float x) {
  const float k = 0.7978845608028654f;
  float t = tanhf(k * (x + 0.044715f * x * x * x));
  return 0.5f * x * (1.0f + t);
}

// CDNA5 async copy: global -> LDS (16B per lane), tracked by ASYNCcnt.
__device__ __forceinline__ void async_copy_b128(void* lds_dst,
                                                const void* gsrc) {
  unsigned lds_off = (unsigned)(uintptr_t)lds_dst;  // low 32 bits = LDS offset
  asm volatile("global_load_async_to_lds_b128 %0, %1, off"
               :
               : "v"(lds_off), "v"(gsrc)
               : "memory");
}
__device__ __forceinline__ void wait_async_le8() {
  asm volatile("s_wait_asynccnt 0x8" ::: "memory");
}
__device__ __forceinline__ void wait_async_0() {
  asm volatile("s_wait_asynccnt 0x0" ::: "memory");
}

// ---------------------------------------------------------------------------
// Weight convert fp32 [K x Nn] row-major  ->  f16 transposed [Nn x K]
// ---------------------------------------------------------------------------
__global__ void cvt_transpose_kernel(const float* __restrict__ W,
                                     half_t* __restrict__ Wt, int K, int Nn) {
  int e = blockIdx.x * 256 + threadIdx.x;
  if (e >= K * Nn) return;
  int n = e / K, k = e - n * K;
  Wt[e] = (half_t)W[(size_t)k * Nn + n];
}

// ---------------------------------------------------------------------------
// Embed: h = feats @ w_in + b_in + positional-embedding concat
// ---------------------------------------------------------------------------
__global__ void embed_kernel(const float* __restrict__ feats,
                             const int* __restrict__ coords,
                             const float* __restrict__ ex,
                             const float* __restrict__ ey,
                             const float* __restrict__ ez,
                             const float* __restrict__ w_in,
                             const float* __restrict__ b_in,
                             float* __restrict__ h) {
  int idx = blockIdx.x * 256 + threadIdx.x;
  if (idx >= N_TOK * C_DIM) return;
  int n = idx >> 9, c = idx & 511;
  float acc = b_in[c];
#pragma unroll
  for (int t = 0; t < 8; ++t) acc += feats[n * 8 + t] * w_in[t * C_DIM + c];
  float pe;
  if (c < 170)       pe = ex[coords[n * 4 + 1] * 170 + c];
  else if (c < 340)  pe = ey[coords[n * 4 + 2] * 170 + (c - 170)];
  else               pe = ez[coords[n * 4 + 3] * 172 + (c - 340)];
  h[idx] = acc + pe;
}

// ---------------------------------------------------------------------------
// LayerNorm (one 256-thread block per row of 512) -> f16 output
// g == nullptr means gamma=1, beta=0 (final LN)
// ---------------------------------------------------------------------------
__global__ __launch_bounds__(256) void ln_kernel(const float* __restrict__ x,
                                                 const float* __restrict__ g,
                                                 const float* __restrict__ b,
                                                 half_t* __restrict__ out,
                                                 float eps) {
  __shared__ float red[256];
  int row = blockIdx.x, tid = threadIdx.x;
  const float* xr = x + (size_t)row * C_DIM;
  float x0 = xr[tid], x1 = xr[tid + 256];
  red[tid] = x0 + x1;
  __syncthreads();
  for (int s = 128; s > 0; s >>= 1) {
    if (tid < s) red[tid] += red[tid + s];
    __syncthreads();
  }
  float mean = red[0] * (1.0f / 512.0f);
  __syncthreads();
  float d0 = x0 - mean, d1 = x1 - mean;
  red[tid] = d0 * d0 + d1 * d1;
  __syncthreads();
  for (int s = 128; s > 0; s >>= 1) {
    if (tid < s) red[tid] += red[tid + s];
    __syncthreads();
  }
  float rstd = rsqrtf(red[0] * (1.0f / 512.0f) + eps);
  float g0 = 1.f, g1 = 1.f, b0 = 0.f, b1 = 0.f;
  if (g) { g0 = g[tid]; g1 = g[tid + 256]; b0 = b[tid]; b1 = b[tid + 256]; }
  out[(size_t)row * C_DIM + tid]       = (half_t)(d0 * rstd * g0 + b0);
  out[(size_t)row * C_DIM + tid + 256] = (half_t)(d1 * rstd * g1 + b1);
}

// ---------------------------------------------------------------------------
// Generic WMMA GEMM: out[M x Nn] = A16[M x K] * Bt16[Nn x K]^T (+bias)
// Tile: 128x128 per 256-thread block (8 waves, each 32x64), k-step 64.
// Double-buffered LDS, filled with GLOBAL_LOAD_ASYNC_TO_LDS_B128 (ASYNCcnt).
// Epilogues: 0 = split to q/k/vT,  1 = residual add into f32,  2 = GELU->f16
// ---------------------------------------------------------------------------
#define EPI_QKV   0
#define EPI_RESID 1
#define EPI_GELU  2
#define TK 64

template <int EPI>
__global__ __launch_bounds__(256) void gemm_wmma_kernel(
    const half_t* __restrict__ A, const half_t* __restrict__ Bt,
    const float* __restrict__ bias, float* __restrict__ res,
    half_t* __restrict__ oq, half_t* __restrict__ ok, half_t* __restrict__ ovT,
    int M, int Nn, int K) {
  __shared__ alignas(16) half_t As[2][128 * TK];
  __shared__ alignas(16) half_t Bs[2][128 * TK];

  int tid = threadIdx.x;
  int lane = tid & 31, wave = tid >> 5;
  int wm = wave & 3, wn = wave >> 2;
  int bm = blockIdx.y * 128, bn = blockIdx.x * 128;

  // Issue one 128x64 tile of A and B into LDS buffer `buf` (4+4 b128/thread).
  auto stage = [&](int buf, int kt) {
#pragma unroll
    for (int i = 0; i < 4; ++i) {
      int c = tid + i * 256;          // 1024 16B-chunks per matrix tile
      int r = c >> 3, cc = (c & 7) * 8;
      async_copy_b128(&As[buf][r * TK + cc],
                      &A[(size_t)(bm + r) * K + kt + cc]);
      async_copy_b128(&Bs[buf][r * TK + cc],
                      &Bt[(size_t)(bn + r) * K + kt + cc]);
    }
  };

  v8f acc[2][4];
#pragma unroll
  for (int mi = 0; mi < 2; ++mi)
#pragma unroll
    for (int ni = 0; ni < 4; ++ni) acc[mi][ni] = v8f{};

  int ntile = K / TK;
  stage(0, 0);
  if (ntile > 1) stage(1, TK);

  for (int t = 0; t < ntile; ++t) {
    int cur = t & 1;
    // Own loads for tile t complete (tile t+1's 8 loads may stay in flight).
    if (t + 1 < ntile) wait_async_le8();
    else               wait_async_0();
    __syncthreads();  // all waves' tile-t data visible in LDS

#pragma unroll
    for (int ks = 0; ks < TK; ks += 32) {
      v16h af[2], bf[4];
#pragma unroll
      for (int mi = 0; mi < 2; ++mi)
        af[mi] = frag_a(&As[cur][(wm * 32 + mi * 16) * TK + ks], TK, lane);
#pragma unroll
      for (int ni = 0; ni < 4; ++ni)
        bf[ni] = frag_b(&Bs[cur][(wn * 64 + ni * 16) * TK + ks], TK, lane);
#pragma unroll
      for (int mi = 0; mi < 2; ++mi)
#pragma unroll
        for (int ni = 0; ni < 4; ++ni)
          acc[mi][ni] = wmma32(af[mi], bf[ni], acc[mi][ni]);
    }
    __syncthreads();  // everyone done reading buffer `cur`
    if (t + 2 < ntile) stage(cur, (t + 2) * TK);
  }

  // Epilogue (C-layout: VGPR r -> row r + 8*(lane>=16), col = lane&15)
  int hl = lane >> 4, cl = lane & 15;
#pragma unroll
  for (int mi = 0; mi < 2; ++mi) {
#pragma unroll
    for (int ni = 0; ni < 4; ++ni) {
#pragma unroll
      for (int r = 0; r < 8; ++r) {
        int row = bm + wm * 32 + mi * 16 + r + 8 * hl;
        int col = bn + wn * 64 + ni * 16 + cl;
        float v = acc[mi][ni][r] + bias[col];
        if (EPI == EPI_QKV) {
          if (col < 512) {
            oq[(size_t)row * C_DIM + col] = (half_t)v;
          } else if (col < 1024) {
            ok[(size_t)row * C_DIM + (col - 512)] = (half_t)v;
          } else {
            int cc = col - 1024;
            int hh = cc >> 6, dh = cc & 63;
            int w = row >> 9, ws = row & 511;
            ovT[(((size_t)(w * H_NUM + hh) * DH + dh) << 9) + ws] = (half_t)v;
          }
        } else if (EPI == EPI_RESID) {
          size_t o = (size_t)row * Nn + col;
          res[o] = res[o] + v;
        } else {  // EPI_GELU
          oq[(size_t)row * Nn + col] = (half_t)gelu_f(v);
        }
      }
    }
  }
}

// ---------------------------------------------------------------------------
// Flash attention: one wave handles 16 query rows of one (window, head).
// Block = 256 thr = 8 waves = 128 query rows; grid = (WS/128, NW3*H).
// q,k: [N][C] f16 (head-major inside row); vT: [wh][dh][ws] f16.
// ---------------------------------------------------------------------------
__global__ __launch_bounds__(256) void attn_kernel(
    const half_t* __restrict__ q, const half_t* __restrict__ k,
    const half_t* __restrict__ vT, half_t* __restrict__ o) {
  __shared__ alignas(16) half_t Pt[8 * 16 * 32];
  int tid = threadIdx.x, lane = tid & 31, wave = tid >> 5;
  int wh = blockIdx.y;             // w*8 + h
  int w = wh >> 3, hh = wh & 7;
  int qb = blockIdx.x * 128 + wave * 16;
  int tok0 = w * WS + qb;

  const half_t* qbase = q + (size_t)tok0 * C_DIM + hh * DH;
  v16h aq0 = frag_a(qbase, C_DIM, lane);        // dh 0..31
  v16h aq1 = frag_a(qbase + 32, C_DIM, lane);   // dh 32..63

  float rowm[8], rowl[8];
  v8f oacc[4];
#pragma unroll
  for (int r = 0; r < 8; ++r) { rowm[r] = -3.0e38f; rowl[r] = 0.0f; }
#pragma unroll
  for (int t = 0; t < 4; ++t) oacc[t] = v8f{};

  half_t* myP = &Pt[wave * 512];
  const half_t* kwin = k + (size_t)(w * WS) * C_DIM + hh * DH;
  const half_t* vbase = vT + ((size_t)wh * DH) * WS;
  int hl = lane >> 4, cl = lane & 15;

  for (int kc = 0; kc < WS; kc += 32) {
    // Prefetch next chunk's K rows into cache (global_prefetch_b8)
    if (kc + 32 < WS)
      __builtin_prefetch(kwin + (size_t)(kc + 32 + (lane & 15)) * C_DIM, 0, 0);

    // S chunk = Q (16x64) x K_chunk^T (64x32), 2 N-tiles x 2 k-steps
    v8f s[2];
#pragma unroll
    for (int nt = 0; nt < 2; ++nt) {
      v16h b0 = frag_b(kwin + (size_t)(kc + nt * 16) * C_DIM, C_DIM, lane);
      v16h b1 = frag_b(kwin + (size_t)(kc + nt * 16) * C_DIM + 32, C_DIM, lane);
      v8f z = v8f{};
      z = wmma32(aq0, b0, z);
      s[nt] = wmma32(aq1, b1, z);
    }
    // Online softmax per row (rows split by VGPR index + lane half)
#pragma unroll
    for (int r = 0; r < 8; ++r) {
      float v0 = s[0][r] * 0.125f, v1 = s[1][r] * 0.125f;
      float mx = fmaxf(v0, v1);
      mx = fmaxf(mx, __shfl_xor(mx, 8, 32));
      mx = fmaxf(mx, __shfl_xor(mx, 4, 32));
      mx = fmaxf(mx, __shfl_xor(mx, 2, 32));
      mx = fmaxf(mx, __shfl_xor(mx, 1, 32));
      float mnew = fmaxf(rowm[r], mx);
      float f = __expf(rowm[r] - mnew);
      float p0 = __expf(v0 - mnew), p1 = __expf(v1 - mnew);
      float sum = p0 + p1;
      sum += __shfl_xor(sum, 8, 32);
      sum += __shfl_xor(sum, 4, 32);
      sum += __shfl_xor(sum, 2, 32);
      sum += __shfl_xor(sum, 1, 32);
      rowl[r] = rowl[r] * f + sum;
      rowm[r] = mnew;
#pragma unroll
      for (int t = 0; t < 4; ++t) oacc[t][r] *= f;
      int rr = r + 8 * hl;
      myP[rr * 32 + cl]      = (half_t)p0;
      myP[rr * 32 + 16 + cl] = (half_t)p1;
    }
    __syncthreads();
    // O += P (16x32) x V_chunk (32x64)
    v16h pf = frag_a(myP, 32, lane);
#pragma unroll
    for (int t = 0; t < 4; ++t) {
      v16h bv = frag_b(vbase + (size_t)(t * 16) * WS + kc, WS, lane);
      oacc[t] = wmma32(pf, bv, oacc[t]);
    }
    __syncthreads();
  }

#pragma unroll
  for (int t = 0; t < 4; ++t)
#pragma unroll
    for (int r = 0; r < 8; ++r) {
      int rr = r + 8 * hl;
      int col = hh * DH + t * 16 + cl;
      o[(size_t)(tok0 + rr) * C_DIM + col] = (half_t)(oacc[t][r] / rowl[r]);
    }
}

// ---------------------------------------------------------------------------
// Output projection 512 -> 56 (tiny FLOPs; plain VALU dot products)
// ---------------------------------------------------------------------------
__global__ void outproj_kernel(const half_t* __restrict__ hf,
                               const float* __restrict__ w,
                               const float* __restrict__ b,
                               float* __restrict__ out) {
  int idx = blockIdx.x * 256 + threadIdx.x;
  if (idx >= N_TOK * OUTC) return;
  int n = idx / OUTC, j = idx - n * OUTC;
  const half_t* hr = hf + (size_t)n * C_DIM;
  float acc = b[j];
  for (int c = 0; c < C_DIM; ++c) acc += (float)hr[c] * w[c * OUTC + j];
  out[idx] = acc;
}

// ---------------------------------------------------------------------------
// Postprocess -> (_xyz, _fdc, _scal, _rot, _opa) concatenated flat
// ---------------------------------------------------------------------------
__global__ void postproc_kernel(const float* __restrict__ o,
                                const int* __restrict__ coords,
                                const float* __restrict__ pert,
                                float* __restrict__ out) {
  int idx = blockIdx.x * 256 + threadIdx.x;  // N * NG
  if (idx >= N_TOK * NG) return;
  int n = idx >> 2, g = idx & 3;
  const float* row = o + (size_t)n * OUTC;
  float* xyz  = out;
  float* fdc  = out + 49152;
  float* scal = out + 98304;
  float* rot  = out + 147456;
  float* opa  = out + 212992;
  const float sc = (1.0f / 16.0f) * 0.5f * 0.015625f;
#pragma unroll
  for (int d = 0; d < 3; ++d) {
    float base = ((float)coords[n * 4 + 1 + d] + 0.5f) / 16.0f;
    float off = tanhf(row[g * 3 + d] + pert[g * 3 + d]) * sc;
    xyz[(size_t)idx * 3 + d]  = base + off;
    fdc[(size_t)idx * 3 + d]  = row[12 + g * 3 + d];
    scal[(size_t)idx * 3 + d] = row[24 + g * 3 + d];
  }
#pragma unroll
  for (int d = 0; d < 4; ++d) rot[(size_t)idx * 4 + d] = row[36 + g * 4 + d];
  opa[idx] = row[52 + g];
}

// ---------------------------------------------------------------------------
// Host launcher
// ---------------------------------------------------------------------------
extern "C" void kernel_launch(void* const* d_in, const int* in_sizes, int n_in,
                              void* d_out, int out_size, void* d_ws,
                              size_t ws_size, hipStream_t stream) {
  const float* feats  = (const float*)d_in[0];
  const int*   coords = (const int*)d_in[1];
  const float* emb_x  = (const float*)d_in[2];
  const float* emb_y  = (const float*)d_in[3];
  const float* emb_z  = (const float*)d_in[4];
  const float* w_in   = (const float*)d_in[5];
  const float* b_in   = (const float*)d_in[6];
  const float* ln1_g  = (const float*)d_in[7];
  const float* ln1_b  = (const float*)d_in[8];
  const float* w_qkv  = (const float*)d_in[9];
  const float* b_qkv  = (const float*)d_in[10];
  const float* w_o    = (const float*)d_in[11];
  const float* b_o    = (const float*)d_in[12];
  const float* ln2_g  = (const float*)d_in[13];
  const float* ln2_b  = (const float*)d_in[14];
  const float* w_m1   = (const float*)d_in[15];
  const float* b_m1   = (const float*)d_in[16];
  const float* w_m2   = (const float*)d_in[17];
  const float* b_m2   = (const float*)d_in[18];
  const float* w_out  = (const float*)d_in[19];
  const float* b_out  = (const float*)d_in[20];
  const float* pert   = (const float*)d_in[21];

  // Workspace carve
  char* ws = (char*)d_ws;
  auto carve = [&](size_t bytes) -> void* {
    void* p = (void*)ws;
    ws += (bytes + 255) & ~(size_t)255;
    return p;
  };
  float*  h    = (float*)carve((size_t)N_TOK * C_DIM * 4);
  half_t* a16  = (half_t*)carve((size_t)N_TOK * C_DIM * 2);   // LN out (reused)
  half_t* q16  = (half_t*)carve((size_t)N_TOK * C_DIM * 2);
  half_t* k16  = (half_t*)carve((size_t)N_TOK * C_DIM * 2);
  half_t* vT   = (half_t*)carve((size_t)N_TOK * C_DIM * 2);
  half_t* o16  = (half_t*)carve((size_t)N_TOK * C_DIM * 2);
  half_t* g16  = (half_t*)carve((size_t)N_TOK * MLP_D * 2);
  float*  o56  = (float*)carve((size_t)N_TOK * OUTC * 4);
  half_t* qkvT = (half_t*)carve((size_t)L_NUM * 3 * C_DIM * C_DIM * 2);
  half_t* woT  = (half_t*)carve((size_t)L_NUM * C_DIM * C_DIM * 2);
  half_t* m1T  = (half_t*)carve((size_t)L_NUM * C_DIM * MLP_D * 2);
  half_t* m2T  = (half_t*)carve((size_t)L_NUM * MLP_D * C_DIM * 2);
  (void)ws_size; (void)n_in; (void)in_sizes; (void)out_size;

  // --- Weight conversion (fp32 [K x N] -> f16 [N x K]) ---
  for (int l = 0; l < L_NUM; ++l) {
    {
      int elts = C_DIM * 3 * C_DIM;
      cvt_transpose_kernel<<<(elts + 255) / 256, 256, 0, stream>>>(
          w_qkv + (size_t)l * elts, qkvT + (size_t)l * elts, C_DIM, 3 * C_DIM);
    }
    {
      int elts = C_DIM * C_DIM;
      cvt_transpose_kernel<<<(elts + 255) / 256, 256, 0, stream>>>(
          w_o + (size_t)l * elts, woT + (size_t)l * elts, C_DIM, C_DIM);
    }
    {
      int elts = C_DIM * MLP_D;
      cvt_transpose_kernel<<<(elts + 255) / 256, 256, 0, stream>>>(
          w_m1 + (size_t)l * elts, m1T + (size_t)l * elts, C_DIM, MLP_D);
      cvt_transpose_kernel<<<(elts + 255) / 256, 256, 0, stream>>>(
          w_m2 + (size_t)l * elts, m2T + (size_t)l * elts, MLP_D, C_DIM);
    }
  }

  // --- Embed ---
  embed_kernel<<<(N_TOK * C_DIM + 255) / 256, 256, 0, stream>>>(
      feats, coords, emb_x, emb_y, emb_z, w_in, b_in, h);

  // --- Transformer layers ---
  for (int l = 0; l < L_NUM; ++l) {
    ln_kernel<<<N_TOK, 256, 0, stream>>>(h, ln1_g + l * C_DIM,
                                         ln1_b + l * C_DIM, a16, 1e-6f);
    // QKV: [4096x512] x [512x1536]
    gemm_wmma_kernel<EPI_QKV><<<dim3(12, 32), 256, 0, stream>>>(
        a16, qkvT + (size_t)l * 3 * C_DIM * C_DIM, b_qkv + l * 3 * C_DIM,
        nullptr, q16, k16, vT, N_TOK, 3 * C_DIM, C_DIM);
    // Attention
    attn_kernel<<<dim3(WS / 128, NW3 * H_NUM), 256, 0, stream>>>(q16, k16, vT,
                                                                 o16);
    // O-proj + residual into h
    gemm_wmma_kernel<EPI_RESID><<<dim3(4, 32), 256, 0, stream>>>(
        o16, woT + (size_t)l * C_DIM * C_DIM, b_o + l * C_DIM, h, nullptr,
        nullptr, nullptr, N_TOK, C_DIM, C_DIM);
    // LN2
    ln_kernel<<<N_TOK, 256, 0, stream>>>(h, ln2_g + l * C_DIM,
                                         ln2_b + l * C_DIM, a16, 1e-6f);
    // MLP1 + GELU: [4096x512] x [512x2048] -> f16
    gemm_wmma_kernel<EPI_GELU><<<dim3(16, 32), 256, 0, stream>>>(
        a16, m1T + (size_t)l * C_DIM * MLP_D, b_m1 + l * MLP_D, nullptr, g16,
        nullptr, nullptr, N_TOK, MLP_D, C_DIM);
    // MLP2 + residual into h: [4096x2048] x [2048x512]
    gemm_wmma_kernel<EPI_RESID><<<dim3(4, 32), 256, 0, stream>>>(
        g16, m2T + (size_t)l * MLP_D * C_DIM, b_m2 + l * C_DIM, h, nullptr,
        nullptr, nullptr, N_TOK, C_DIM, MLP_D);
  }

  // --- Final LN (gamma=1, beta=0, eps=1e-5) ---
  ln_kernel<<<N_TOK, 256, 0, stream>>>(h, nullptr, nullptr, a16, 1e-5f);

  // --- Output projection + postprocess ---
  outproj_kernel<<<(N_TOK * OUTC + 255) / 256, 256, 0, stream>>>(a16, w_out,
                                                                 b_out, o56);
  postproc_kernel<<<(N_TOK * NG + 255) / 256, 256, 0, stream>>>(
      o56, coords, pert, (float*)d_out);
}
